// GNet_13537736917056
// MI455X (gfx1250) — matrix operand order repaired and checked by reading, not compiled, not run on hardware
//
#include <hip/hip_runtime.h>
#include <math.h>

// ---------------- types ----------------
typedef __attribute__((ext_vector_type(16))) __bf16 v16bf;
typedef __attribute__((ext_vector_type(8)))  float  v8f;
typedef unsigned int u32x4 __attribute__((ext_vector_type(4)));

#define B_   128
#define L_   512
#define H_   512
#define I_   96
#define OC_  16
#define CC_  64
#define NBLK 32

// ---------------- helpers ----------------
__device__ __forceinline__ unsigned short f2bf(float f) {
    unsigned int u = __float_as_uint(f);
    unsigned int r = u + 0x7FFFu + ((u >> 16) & 1u);   // round-to-nearest-even
    return (unsigned short)(r >> 16);
}

// A fragment (16x32 bf16): two 16B chunks at p and p+16 elements
__device__ __forceinline__ v16bf ldA(const unsigned short* p) {
    union { v16bf v; u32x4 q[2]; } u;
    u.q[0] = *(const u32x4*)p;
    u.q[1] = *(const u32x4*)(p + 16);
    return u.v;
}

// B fragment: one lane's 16 contiguous bf16 (two 16B chunks at p, p+8)
__device__ __forceinline__ v16bf ldB16(const unsigned short* p) {
    union { v16bf v; u32x4 q[2]; } u;
    u.q[0] = *(const u32x4*)p;
    u.q[1] = *(const u32x4*)(p + 8);
    return u.v;
}

__device__ __forceinline__ v8f wmma_bf16(v16bf a, v16bf b, v8f c) {
    return __builtin_amdgcn_wmma_f32_16x16x32_bf16(false, a, false, b, (short)0, c, false, false);
}

__device__ __forceinline__ v8f vfill(float x) {
    v8f v;
#pragma unroll
    for (int i = 0; i < 8; ++i) v[i] = x;
    return v;
}

__device__ __forceinline__ float sigf(float x)  { return 1.0f / (1.0f + __expf(-x)); }
__device__ __forceinline__ float tanhf_(float x){ float e = __expf(2.0f * x); return 1.0f - 2.0f / (e + 1.0f); }

__device__ __forceinline__ void gridbar(unsigned int* cnt, unsigned int phase) {
    __syncthreads();
    if (threadIdx.x == 0) {
        __threadfence();
        atomicAdd(cnt, 1u);
        const unsigned int target = phase * (unsigned int)NBLK;
        while (atomicAdd(cnt, 0u) < target) __builtin_amdgcn_s_sleep(2);
    }
    __syncthreads();
    __threadfence();
}

// ---------------- kernel 1: RevIN stats + normalize + concat (bf16) ----------------
__global__ void revin_kernel(const float* __restrict__ cov, const float* __restrict__ tre,
                             const float* __restrict__ outc,
                             const float* w_t, const float* b_t, const float* w_o, const float* b_o,
                             const float* w_c, const float* b_c,
                             unsigned short* __restrict__ xn,
                             float* om, float* osd, float* cm, float* csd) {
    int b = blockIdx.x;
    int ch = threadIdx.x;
    if (ch >= I_) return;
    const float* src; int C, c; float w, bb;
    if (ch < 16)      { src = tre;  C = 16; c = ch;      w = w_t[c]; bb = b_t[c]; }
    else if (ch < 32) { src = outc; C = 16; c = ch - 16; w = w_o[c]; bb = b_o[c]; }
    else              { src = cov;  C = 64; c = ch - 32; w = w_c[c]; bb = b_c[c]; }
    const float* p = src + (size_t)b * L_ * C + c;
    float s1 = 0.0f, s2 = 0.0f;
    for (int l = 0; l < L_; ++l) { float v = p[(size_t)l * C]; s1 += v; s2 += v * v; }
    float m   = s1 * (1.0f / (float)L_);
    float var = (s2 - s1 * m) * (1.0f / (float)(L_ - 1));
    var = var < 0.0f ? 0.0f : var;
    float sd = sqrtf(var) + 1e-5f;
    float inv = 1.0f / sd;
    for (int l = 0; l < L_; ++l) {
        float v = (p[(size_t)l * C] - m) * inv * w + bb;
        xn[((size_t)b * L_ + l) * I_ + ch] = f2bf(v);
    }
    if (ch >= 16 && ch < 32) { om[b * 16 + c] = m; osd[b * 16 + c] = sd; }
    else if (ch >= 32)       { cm[b * 64 + c] = m; csd[b * 64 + c] = sd; }
}

// ---------------- kernel 2: pack weight (N,K) f32 -> WMMA B-tiles bf16 ----------------
// packed[((ntile*KT + kt)*32 + lane)*16 + j] = bf16(W[ntile*16 + (lane&15), kt*32 + (j>>3)*16 + (lane>>4)*8 + (j&7)])
__global__ void pack_w(const float* __restrict__ W, unsigned short* __restrict__ out, int N, int K) {
    int idx = blockIdx.x * blockDim.x + threadIdx.x;
    if (idx >= N * K) return;
    int KT   = K >> 5;
    int j    = idx & 15;
    int lane = (idx >> 4) & 31;
    int tile = idx >> 9;
    int kt = tile % KT, nt = tile / KT;
    int n = (nt << 4) + (lane & 15);
    int k = (kt << 5) + ((j >> 3) << 4) + ((lane >> 4) << 3) + (j & 7);
    out[idx] = f2bf(W[(size_t)n * K + k]);
}

// ---------------- kernel 3: persistent 2-layer GRU + output projection ----------------
// LDS: per-WG weight slices staged once, reused for all 512 steps.
__global__ __launch_bounds__(256, 1)
void gru_persistent(const unsigned short* __restrict__ xn,
                    unsigned short* __restrict__ h0buf, unsigned short* __restrict__ h1buf,
                    const unsigned short* __restrict__ wih0p, const unsigned short* __restrict__ whh0p,
                    const unsigned short* __restrict__ wih1p, const unsigned short* __restrict__ whh1p,
                    const unsigned short* __restrict__ woutp,
                    const float* __restrict__ bih0, const float* __restrict__ bhh0,
                    const float* __restrict__ bih1, const float* __restrict__ bhh1,
                    const float* __restrict__ b_out,
                    const float* __restrict__ w_o, const float* __restrict__ b_o,
                    const float* __restrict__ w_c, const float* __restrict__ b_c,
                    const float* __restrict__ om, const float* __restrict__ osd,
                    const float* __restrict__ cm, const float* __restrict__ csd,
                    unsigned int* __restrict__ bar, float* __restrict__ out) {
    // 153 KB of the 320 KB WGP LDS: [gate][kt][32 lanes][16 bf16] tiles
    __shared__ unsigned short sWhh0[3 * 16 * 512];
    __shared__ unsigned short sWih1[3 * 16 * 512];
    __shared__ unsigned short sWhh1[3 * 16 * 512];
    __shared__ unsigned short sWih0[3 * 3  * 512];

    const int wg   = blockIdx.x;            // 0..31 : owns H columns [16*wg, 16*wg+16)
    const int lane = threadIdx.x & 31;
    const int wave = threadIdx.x >> 5;      // 0..7 : owns M rows [16*wave, 16*wave+16)
    const int rm   = wave << 4;
    const int nlo  = lane & 15;
    const int col  = (wg << 4) + nlo;
    const int hi8  = (lane >> 4) << 3;

    // ---- stage weight slices global -> LDS (once) ----
    {
        u32x4* d0 = (u32x4*)sWhh0; const u32x4* g0 = (const u32x4*)whh0p;
        u32x4* d1 = (u32x4*)sWih1; const u32x4* g1 = (const u32x4*)wih1p;
        u32x4* d2 = (u32x4*)sWhh1; const u32x4* g2 = (const u32x4*)whh1p;
        for (int i = threadIdx.x; i < 3 * 2048; i += 256) {   // 16 tiles * 512 ushorts = 1024 uint4 per gate
            int g = i >> 11, r = i & 2047;
            int src = ((g * 32 + wg) << 10) + r;              // (g*32+wg)*16*512/8
            d0[i] = g0[src];
            d1[i] = g1[src];
            d2[i] = g2[src];
        }
        u32x4* d3 = (u32x4*)sWih0; const u32x4* g3 = (const u32x4*)wih0p;
        for (int i = threadIdx.x; i < 3 * 192; i += 256) {    // 3 tiles * 512 ushorts = 192 uint4 per gate
            int g = i / 192, r = i - g * 192;
            d3[i] = g3[(g * 32 + wg) * 192 + r];
        }
    }
    __syncthreads();

    // ---- loop-invariant hoists ----
    const float b0r = bih0[col] + bhh0[col];
    const float b0z = bih0[512 + col] + bhh0[512 + col];
    const float b0i = bih0[1024 + col];
    const float b0h = bhh0[1024 + col];
    const float b1r = bih1[col] + bhh1[col];
    const float b1z = bih1[512 + col] + bhh1[512 + col];
    const float b1i = bih1[1024 + col];
    const float b1h = bhh1[1024 + col];

    const int aoff  = ((rm + nlo) << 9) + hi8;               // (row)*512 + k-chunk, for h buffers
    const int xbase = (rm + nlo) * L_ * I_ + hi8;            // + l*96 + kt*32, for x_n

    // output-projection invariants (uniform per block: wg0 -> outcomes, wg1..4 -> covariates)
    float  o_bias = 0.0f, o_bb = 0.0f, o_iw = 1.0f;
    float  o_m[8], o_s[8];
    size_t o_base[8];
    size_t o_stride = 0;
    if (wg < 5) {
        const int oc = (wg << 4) + nlo;                      // 0..79
        o_bias = b_out[oc];
        if (wg == 0) {
            o_bb = b_o[oc]; o_iw = 1.0f / w_o[oc]; o_stride = OC_;
#pragma unroll
            for (int i = 0; i < 8; ++i) {
                int b = rm + hi8 + i;
                o_s[i] = osd[b * 16 + oc]; o_m[i] = om[b * 16 + oc];
                o_base[i] = (size_t)b * L_ * OC_ + oc;
            }
        } else {
            const int c = oc - OC_;
            o_bb = b_c[c]; o_iw = 1.0f / w_c[c]; o_stride = CC_;
#pragma unroll
            for (int i = 0; i < 8; ++i) {
                int b = rm + hi8 + i;
                o_s[i] = csd[b * 64 + c]; o_m[i] = cm[b * 64 + c];
                o_base[i] = (size_t)B_ * L_ * OC_ + (size_t)b * L_ * CC_ + c;
            }
        }
    }

    v8f h0s = vfill(0.0f);
    v8f h1s = vfill(0.0f);
    unsigned int phase = 0;

    for (int l = 0; l < L_; ++l) {
        const int cur = l & 1, prv = cur ^ 1;
        unsigned short*       h0c = h0buf + cur * (B_ * H_);
        const unsigned short* h0p = h0buf + prv * (B_ * H_);
        unsigned short*       h1c = h1buf + cur * (B_ * H_);
        const unsigned short* h1p = h1buf + prv * (B_ * H_);

        // ===== layer 0 gates =====
        v8f rA  = vfill(b0r);
        v8f zA  = vfill(b0z);
        v8f niA = vfill(b0i);
        v8f nhA = vfill(b0h);
        {
            const unsigned short* xp = xn + xbase + l * I_;
#pragma unroll
            for (int kt = 0; kt < 3; ++kt) {                 // K = 96 (input)
                v16bf a = ldA(xp + (kt << 5));
                rA  = wmma_bf16(a, ldB16(sWih0 + (((0 * 3 + kt) << 5) + lane) * 16), rA);
                zA  = wmma_bf16(a, ldB16(sWih0 + (((1 * 3 + kt) << 5) + lane) * 16), zA);
                niA = wmma_bf16(a, ldB16(sWih0 + (((2 * 3 + kt) << 5) + lane) * 16), niA);
            }
        }
#pragma unroll 2
        for (int kt = 0; kt < 16; ++kt) {                    // K = 512 (hidden)
            v16bf a = ldA(h0p + aoff + (kt << 5));
            rA  = wmma_bf16(a, ldB16(sWhh0 + (((0 * 16 + kt) << 5) + lane) * 16), rA);
            zA  = wmma_bf16(a, ldB16(sWhh0 + (((1 * 16 + kt) << 5) + lane) * 16), zA);
            nhA = wmma_bf16(a, ldB16(sWhh0 + (((2 * 16 + kt) << 5) + lane) * 16), nhA);
        }
#pragma unroll
        for (int i = 0; i < 8; ++i) {
            float r = sigf(rA[i]);
            float z = sigf(zA[i]);
            float n = tanhf_(niA[i] + r * nhA[i]);
            float h = (1.0f - z) * n + z * h0s[i];
            h0s[i] = h;
            h0c[(rm + hi8 + i) * H_ + col] = f2bf(h);
        }
        gridbar(bar, ++phase);                               // h0(cur) visible everywhere

        // ===== layer 1 gates =====
        rA  = vfill(b1r);
        zA  = vfill(b1z);
        niA = vfill(b1i);
        nhA = vfill(b1h);
#pragma unroll 2
        for (int kt = 0; kt < 16; ++kt) {
            v16bf a = ldA(h0c + aoff + (kt << 5));
            rA  = wmma_bf16(a, ldB16(sWih1 + (((0 * 16 + kt) << 5) + lane) * 16), rA);
            zA  = wmma_bf16(a, ldB16(sWih1 + (((1 * 16 + kt) << 5) + lane) * 16), zA);
            niA = wmma_bf16(a, ldB16(sWih1 + (((2 * 16 + kt) << 5) + lane) * 16), niA);
        }
#pragma unroll 2
        for (int kt = 0; kt < 16; ++kt) {
            v16bf a = ldA(h1p + aoff + (kt << 5));
            rA  = wmma_bf16(a, ldB16(sWhh1 + (((0 * 16 + kt) << 5) + lane) * 16), rA);
            zA  = wmma_bf16(a, ldB16(sWhh1 + (((1 * 16 + kt) << 5) + lane) * 16), zA);
            nhA = wmma_bf16(a, ldB16(sWhh1 + (((2 * 16 + kt) << 5) + lane) * 16), nhA);
        }
#pragma unroll
        for (int i = 0; i < 8; ++i) {
            float r = sigf(rA[i]);
            float z = sigf(zA[i]);
            float n = tanhf_(niA[i] + r * nhA[i]);
            float h = (1.0f - z) * n + z * h1s[i];
            h1s[i] = h;
            h1c[(rm + hi8 + i) * H_ + col] = f2bf(h);
        }
        gridbar(bar, ++phase);                               // h1(cur) visible everywhere

        // ===== output projection (wg 0..4) + RevIN inverse =====
        if (wg < 5) {
            v8f acc = vfill(o_bias);
            const unsigned short* wop = woutp + (wg << 13);  // wg*16*512
#pragma unroll 2
            for (int kt = 0; kt < 16; ++kt) {
                v16bf a = ldA(h1c + aoff + (kt << 5));
                acc = wmma_bf16(a, ldB16(wop + ((kt << 5) + lane) * 16), acc);
            }
#pragma unroll
            for (int i = 0; i < 8; ++i) {
                out[o_base[i] + (size_t)l * o_stride] = (acc[i] - o_bb) * o_iw * o_s[i] + o_m[i];
            }
        }
    }
}

// ---------------- host ----------------
extern "C" void kernel_launch(void* const* d_in, const int* in_sizes, int n_in,
                              void* d_out, int out_size, void* d_ws, size_t ws_size,
                              hipStream_t stream) {
    const float* cov  = (const float*)d_in[0];
    const float* tre  = (const float*)d_in[1];
    const float* outc = (const float*)d_in[2];
    const float* Wih0 = (const float*)d_in[3];
    const float* Whh0 = (const float*)d_in[4];
    const float* bih0 = (const float*)d_in[5];
    const float* bhh0 = (const float*)d_in[6];
    const float* Wih1 = (const float*)d_in[7];
    const float* Whh1 = (const float*)d_in[8];
    const float* bih1 = (const float*)d_in[9];
    const float* bhh1 = (const float*)d_in[10];
    const float* Wout = (const float*)d_in[11];
    const float* bout = (const float*)d_in[12];
    const float* w_t  = (const float*)d_in[13];
    const float* b_t  = (const float*)d_in[14];
    const float* w_o  = (const float*)d_in[15];
    const float* b_o  = (const float*)d_in[16];
    const float* w_c  = (const float*)d_in[17];
    const float* b_c  = (const float*)d_in[18];
    float* out = (float*)d_out;

    char* ws = (char*)d_ws;
    size_t off = 0;
    auto carve = [&](size_t bytes) -> char* {
        char* p = ws + off;
        off = (off + bytes + 255) & ~(size_t)255;
        return p;
    };
    unsigned int*   bar   = (unsigned int*)carve(256);
    unsigned short* h0buf = (unsigned short*)carve(2 * B_ * H_ * sizeof(unsigned short));
    unsigned short* h1buf = (unsigned short*)carve(2 * B_ * H_ * sizeof(unsigned short));
    size_t zeroBytes = off;                    // bar + h0 + h1 must start at zero
    unsigned short* xn    = (unsigned short*)carve((size_t)B_ * L_ * I_ * 2);
    unsigned short* wih0p = (unsigned short*)carve((size_t)1536 * 96  * 2);
    unsigned short* whh0p = (unsigned short*)carve((size_t)1536 * 512 * 2);
    unsigned short* wih1p = (unsigned short*)carve((size_t)1536 * 512 * 2);
    unsigned short* whh1p = (unsigned short*)carve((size_t)1536 * 512 * 2);
    unsigned short* woutp = (unsigned short*)carve((size_t)80   * 512 * 2);
    float* om  = (float*)carve((size_t)B_ * 16 * 4);
    float* osd = (float*)carve((size_t)B_ * 16 * 4);
    float* cm  = (float*)carve((size_t)B_ * 64 * 4);
    float* csd = (float*)carve((size_t)B_ * 64 * 4);

    hipMemsetAsync(d_ws, 0, zeroBytes, stream);

    revin_kernel<<<B_, 96, 0, stream>>>(cov, tre, outc, w_t, b_t, w_o, b_o, w_c, b_c,
                                        xn, om, osd, cm, csd);

    pack_w<<<(1536 * 96  + 255) / 256, 256, 0, stream>>>(Wih0, wih0p, 1536, 96);
    pack_w<<<(1536 * 512 + 255) / 256, 256, 0, stream>>>(Whh0, whh0p, 1536, 512);
    pack_w<<<(1536 * 512 + 255) / 256, 256, 0, stream>>>(Wih1, wih1p, 1536, 512);
    pack_w<<<(1536 * 512 + 255) / 256, 256, 0, stream>>>(Whh1, whh1p, 1536, 512);
    pack_w<<<(80   * 512 + 255) / 256, 256, 0, stream>>>(Wout, woutp, 80, 512);

    gru_persistent<<<NBLK, 256, 0, stream>>>(xn, h0buf, h1buf,
                                             wih0p, whh0p, wih1p, whh1p, woutp,
                                             bih0, bhh0, bih1, bhh1, bout,
                                             w_o, b_o, w_c, b_c,
                                             om, osd, cm, csd, bar, out);
}
